// Classifier_with_Augmentation_77481210020196
// MI455X (gfx1250) — matrix operand order repaired and checked by reading, not compiled, not run on hardware
//
#include <hip/hip_runtime.h>
#include <hip/hip_bf16.h>

// ---------------------------------------------------------------------------
// CDNA5 (gfx1250) implementation of Classifier_with_Augmentation.
//   - conv layers: implicit GEMM via v_wmma_f32_16x16x32_f16 (wave32),
//     geometry templated, LDS tiles in WMMA fragment layout (ds_*_b128),
//     weight tiles DMA'd by the Tensor Data Mover (tensor_load_to_lds +
//     s_wait_tensorcnt) overlapped with activation staging.
//     4 waves/block, each wave computes 32x32 of output = 4 independent
//     WMMAs per K-chunk (good co-issue; f16 WMMA RAW hazard hidden).
//   - APPM window scoring + NMS, bilinear crop-resize, pool+head: VALU kernels
// ---------------------------------------------------------------------------

typedef __attribute__((ext_vector_type(16))) _Float16 v16h;
typedef __attribute__((ext_vector_type(8)))  _Float16 v8h;
typedef __attribute__((ext_vector_type(8)))  float    v8f;
typedef __attribute__((ext_vector_type(4)))  unsigned int u32x4;
typedef __attribute__((ext_vector_type(8)))  int          i32x8;
typedef __attribute__((ext_vector_type(4)))  int          i32x4;

#if defined(__has_builtin)
#if __has_builtin(__builtin_amdgcn_tensor_load_to_lds) && \
    __has_builtin(__builtin_amdgcn_s_wait_tensorcnt)
#define HAVE_TDM 1
#endif
#endif
#ifndef HAVE_TDM
#define HAVE_TDM 0
#endif

#if HAVE_TDM
#warning "CDNA5 TDM path: ENABLED (__builtin_amdgcn_tensor_load_to_lds)"
#else
#warning "CDNA5 TDM path: DISABLED (builtin unavailable; sync staging fallback)"
#endif

#if HAVE_TDM
// Issue one TDM 2D-tile load: 64 rows x 32 f16 from [rows][Ktot] f16 tensor
// into LDS at ldsAddr (rows packed contiguously, 64B each). OOB columns
// (k >= Ktot) are zero-filled by the TDM per the D# tensor_dim0 bound.
__device__ __forceinline__ void tdm_load_btile(const _Float16* gbase,
                                               unsigned ldsAddr,
                                               int Ktot, int Co) {
  unsigned long long ga = (unsigned long long)gbase;
  u32x4 g0;
  g0[0] = 1u;                                        // count=1 (valid), user mode
  g0[1] = ldsAddr;                                   // lds_addr (bytes)
  g0[2] = (unsigned)(ga & 0xffffffffu);              // global_addr[31:0]
  g0[3] = (unsigned)((ga >> 32) & 0x01ffffffu)       // global_addr[56:32]
          | (2u << 30);                              // type=2 ("image")
  i32x8 g1;
  g1[0] = (int)(1u << 16);                           // data_size=1 -> 2 bytes
  g1[1] = (int)(((unsigned)Ktot & 0xffffu) << 16);   // tensor_dim0[15:0]
  g1[2] = (int)((((unsigned)Ktot >> 16) & 0xffffu)   // tensor_dim0[31:16]
          | (((unsigned)Co & 0xffffu) << 16));       // tensor_dim1[15:0]
  g1[3] = (int)((((unsigned)Co >> 16) & 0xffffu)     // tensor_dim1[31:16]
          | (32u << 16));                            // tile_dim0 = 32
  g1[4] = 64;                                        // tile_dim1 = 64 rows
  g1[5] = Ktot;                                      // tensor_dim0_stride[31:0]
  g1[6] = 0;
  g1[7] = 0;
  i32x4 z4 = {};
#if __clang_major__ >= 23
  i32x8 z8 = {};
  __builtin_amdgcn_tensor_load_to_lds(g0, g1, z4, z4, z8, 0);
#else
  __builtin_amdgcn_tensor_load_to_lds(g0, g1, z4, z4, 0);
#endif
}
#endif

// -------------------------------- packing ----------------------------------

__global__ void pack_nchw_to_nhwc_f16(const float* __restrict__ in,
                                      _Float16* __restrict__ out,
                                      int B, int C, int H, int W) {
  long total = (long)B * C * H * W;
  long idx = (long)blockIdx.x * blockDim.x + threadIdx.x;
  if (idx >= total) return;
  long t = idx;
  int c = t % C; t /= C;
  int w = t % W; t /= W;
  int h = t % H; t /= H;
  int b = (int)t;
  out[idx] = (_Float16)in[(((long)b * C + c) * H + h) * W + w];
}

// OIHW f32  ->  [Co][K] f16 with K = (ky*3+kx)*Ci + ci   (K-contiguous rows)
__global__ void pack_weights_f16(const float* __restrict__ w,
                                 _Float16* __restrict__ wk,
                                 int Co, int Ci) {
  long total = (long)9 * Ci * Co;
  long idx = (long)blockIdx.x * blockDim.x + threadIdx.x;
  if (idx >= total) return;
  int K = 9 * Ci;
  int k = (int)(idx % K);
  int o = (int)(idx / K);
  int ci = k % Ci;
  int kk = k / Ci;                 // ky*3+kx
  wk[idx] = (_Float16)w[((long)o * Ci + ci) * 9 + kk];
}

// ------------------------- implicit-GEMM conv (WMMA) -----------------------
// stride-2 3x3 conv, SAME padding, bias + ReLU fused.
// GEMM: M = Ho*Wo (per image), N = Co, K = 9*Ci.
// Block = 128 threads = 4 waves; block tile 64(M) x 64(N); K chunk 32.
// Wave (wmr, wnc): each wave owns a 32x32 output tile -> 2 A-fragments x
// 2 B-fragments -> 4 independent WMMAs per chunk on 4 accumulators.
//
// Per chunk: wave0 launches a TDM DMA of the 64x32 weight tile into LDS,
// overlapped with all-thread global staging of the A tile (already in
// fragment layout); after s_wait_tensorcnt + barrier, the raw weight tile is
// permuted LDS->LDS into fragment layout, then fragments are read as 32B
// vectors (2x ds_load_b128 each) and fed to the WMMAs.
//
// Fragment layouts (CDNA5 wave32, 16-bit):
//   A (16x32): lane = (m&15) + ((kloc%16)>=8)*16 ; half = (kloc>=16)*8 + (kloc&7)
//   B (32x16): lane = (n&15) + ((kloc%16)>=8)*16 ; half = (kloc>=16)*8 + (kloc&7)
template <int Hi, int Wi, int Ci, int Ho, int Wo, int Co, int PAD>
__global__ __launch_bounds__(128) void conv3x3_s2_wmma(
    const _Float16* __restrict__ in,   // [B][Hi][Wi][Ci]
    const _Float16* __restrict__ wkT,  // [Co][K]
    const float*    __restrict__ bias, // [Co]
    _Float16*       __restrict__ out)  // [B][M][Co]  (NHWC flattened)
{
  constexpr int M    = Ho * Wo;
  constexpr int Ktot = 9 * Ci;
  constexpr int KCH  = (Ktot + 31) / 32;
  constexpr bool MEXACT = (M % 64) == 0;

  __shared__ v16h aF[4][32];            // A fragments [16-row sub-tile][lane]
  __shared__ v16h bF[4][32];            // B fragments [16-col sub-tile][lane]
#if HAVE_TDM
  __shared__ _Float16 rawB[64 * 32];    // TDM landing tile [64 n][32 k]
#endif

  const int bb   = blockIdx.z;
  const int m0   = blockIdx.x * 64;
  const int n0   = blockIdx.y * 64;
  const int tid  = threadIdx.x;
  const int lane = tid & 31;
  const int wave = tid >> 5;
  const int wmr  = (wave >> 1) * 2;   // first of two 16-row sub-tiles
  const int wnc  = (wave & 1) * 2;    // first of two 16-col sub-tiles

  // ---- per-thread A staging constants (2 octet-units each; hoisted)
  _Float16* aDst[2];
  bool aOK[2];
  int air0[2], aic0[2];
  int akg;
#pragma unroll
  for (int u = 0; u < 2; ++u) {
    const int e  = tid + u * 128;      // 0..255
    const int mi = e >> 2;             // tile row 0..63
    akg = e & 3;                       // K octet (same for both units)
    const int m = m0 + mi;
    aOK[u] = MEXACT || (m < M);
    aDst[u] = (_Float16*)&aF[mi >> 4][(mi & 15) + ((akg & 1) << 4)]
              + ((akg >> 1) << 3);
    const int ho = m / Wo, wo = m % Wo;   // constant divisor
    air0[u] = 2 * ho - PAD;
    aic0[u] = 2 * wo - PAD;
  }
  const _Float16* inB = in + (long)bb * Hi * Wi * Ci;

  // ---- per-thread B rearrange constants (2 octet-units each)
  _Float16* bDst[2];
#if HAVE_TDM
  const _Float16* bSrc[2];
#else
  const _Float16* bRow[2];
#endif
#pragma unroll
  for (int u = 0; u < 2; ++u) {
    const int e   = tid + u * 128;
    const int bni = e >> 2;
    const int bkg = e & 3;
    bDst[u] = (_Float16*)&bF[bni >> 4][(bni & 15) + ((bkg & 1) << 4)]
              + ((bkg >> 1) << 3);
#if HAVE_TDM
    bSrc[u] = rawB + bni * 32 + (bkg << 3);
#else
    bRow[u] = wkT + (long)(n0 + bni) * Ktot + (bkg << 3);
#endif
  }
#if HAVE_TDM
  const unsigned rawBAddr = (unsigned)(unsigned long long)(const void*)rawB;
#endif

  v8f acc00 = {}, acc01 = {}, acc10 = {}, acc11 = {};

  for (int kc = 0; kc < KCH; ++kc) {
    const int k0 = kc * 32;

#if HAVE_TDM
    // TDM: DMA the 64x32 weight tile into LDS (wave0 only; TDM ignores EXEC,
    // so a wave-uniform branch issues exactly one DMA)
    if (wave == 0)
      tdm_load_btile(wkT + (long)n0 * Ktot + k0, rawBAddr, Ktot, Co);
#endif

    // ---------------- stage A tile (global -> LDS, fragment layout)
#pragma unroll
    for (int u = 0; u < 2; ++u) {
      v8h av = {};
      if constexpr (Ci % 32 == 0) {
        // whole 32-chunk lives inside one (ky,kx) slice; contiguous in ci
        const int kk  = k0 / Ci;
        const int ky  = kk / 3, kx = kk % 3;
        const int ci0 = (k0 % Ci) + (akg << 3);
        const int ir = air0[u] + ky, ic = aic0[u] + kx;
        if (aOK[u] && (unsigned)ir < (unsigned)Hi && (unsigned)ic < (unsigned)Wi)
          av = *(const v8h*)(inB + ((long)ir * Wi + ic) * Ci + ci0);
      } else {
        if (aOK[u]) {
#pragma unroll
          for (int j = 0; j < 8; ++j) {
            const int k = k0 + (akg << 3) + j;
            if (k < Ktot) {
              const int ci = k % Ci;
              const int kk = k / Ci;
              const int ky = kk / 3, kx = kk % 3;
              const int ir = air0[u] + ky, ic = aic0[u] + kx;
              if ((unsigned)ir < (unsigned)Hi && (unsigned)ic < (unsigned)Wi)
                av[j] = inB[((long)ir * Wi + ic) * Ci + ci];
            }
          }
        }
      }
      *(v8h*)aDst[u] = av;
    }

#if HAVE_TDM
    if (wave == 0) __builtin_amdgcn_s_wait_tensorcnt(0);
    __syncthreads();   // rawB + aF visible

    // permute raw weight tile LDS->LDS into fragment layout
#pragma unroll
    for (int u = 0; u < 2; ++u) *(v8h*)bDst[u] = *(const v8h*)bSrc[u];
#else
    // stage B tile (global -> LDS, fragment layout)
#pragma unroll
    for (int u = 0; u < 2; ++u) {
      v8h bv = {};
      if constexpr (Ktot % 32 == 0) {
        bv = *(const v8h*)(bRow[u] + k0);
      } else {
        const int kb = (int)((tid + u * 128) & 3) << 3;
#pragma unroll
        for (int j = 0; j < 8; ++j) {
          const int k = k0 + kb + j;
          if (k < Ktot) bv[j] = bRow[u][k - kb];
        }
      }
      *(v8h*)bDst[u] = bv;
    }
#endif
    __syncthreads();

    // ---------------- fragments straight off LDS (2x ds_load_b128 each)
    const v16h af0 = aF[wmr][lane];
    const v16h af1 = aF[wmr + 1][lane];
    const v16h bf0 = bF[wnc][lane];
    const v16h bf1 = bF[wnc + 1][lane];

    // 4 independent WMMAs (no D->A/B RAW between them)
    acc00 = __builtin_amdgcn_wmma_f32_16x16x32_f16(
        false, af0, false, bf0, (short)0, acc00, false, false);
    acc01 = __builtin_amdgcn_wmma_f32_16x16x32_f16(
        false, af0, false, bf1, (short)0, acc01, false, false);
    acc10 = __builtin_amdgcn_wmma_f32_16x16x32_f16(
        false, af1, false, bf0, (short)0, acc10, false, false);
    acc11 = __builtin_amdgcn_wmma_f32_16x16x32_f16(
        false, af1, false, bf1, (short)0, acc11, false, false);

    __syncthreads();
  }

  // ---------------- store D: bias + ReLU, NHWC f16
  // D layout: VGPR v -> M = v + (lane>=16)*8 ; N = lane&15
  // NHWC flattening: out[(bb*M + mrow)*Co + ncol] (no div/mod needed)
  _Float16* outB = out + (long)bb * M * Co;
  const int nb = n0 + wnc * 16 + (lane & 15);
  const float bs0 = bias[nb];
  const float bs1 = bias[nb + 16];
#pragma unroll
  for (int a = 0; a < 2; ++a) {
    const int mbase = m0 + (wmr + a) * 16 + (lane >> 4) * 8;
    const v8f* ac0 = (a == 0) ? &acc00 : &acc10;
    const v8f* ac1 = (a == 0) ? &acc01 : &acc11;
#pragma unroll
    for (int v = 0; v < 8; ++v) {
      const int mrow = mbase + v;
      if (MEXACT || mrow < M) {
        float r0 = (*ac0)[v] + bs0;
        float r1 = (*ac1)[v] + bs1;
        r0 = r0 > 0.f ? r0 : 0.f;
        r1 = r1 > 0.f ? r1 : 0.f;
        outB[(long)mrow * Co + nb]      = (_Float16)r0;
        outB[(long)mrow * Co + nb + 16] = (_Float16)r1;
      }
    }
  }
}

// ------------------------- global-avg-pool + head --------------------------

__global__ __launch_bounds__(256) void pool_head(
    const _Float16* __restrict__ act,  // [B][H][W][512]
    const float* __restrict__ hw,      // [512][200]
    const float* __restrict__ hb,      // [200]
    float* __restrict__ logits,        // [B][200]
    int HW)
{
  __shared__ float feat[512];
  const int b = blockIdx.x;
  const _Float16* p = act + (long)b * HW * 512;
  for (int c = threadIdx.x; c < 512; c += 256) {
    float s = 0.f;
    for (int i = 0; i < HW; ++i) s += (float)p[(long)i * 512 + c];
    feat[c] = s / (float)HW;
  }
  __syncthreads();
  for (int t = threadIdx.x; t < 200; t += 256) {
    float s = hb[t];
    for (int c = 0; c < 512; ++c) s += feat[c] * hw[c * 200 + t];
    logits[(long)b * 200 + t] = s;
  }
}

// ------------------------------ APPM + NMS ---------------------------------

__device__ __forceinline__ void box_of(int w, float* bx) {
  const int rhs[6] = {4, 3, 5, 6, 5, 7};
  const int rws[6] = {4, 5, 3, 6, 7, 5};
  const int off[7] = {0, 121, 241, 361, 442, 522, 602};
  int ri = 0;
  while (ri < 5 && w >= off[ri + 1]) ri++;
  int loc = w - off[ri];
  int cols = 14 - rws[ri] + 1;
  int r = loc / cols, c = loc % cols;
  bx[0] = (float)(r * 32);
  bx[1] = (float)(c * 32);
  bx[2] = (float)((r + rhs[ri]) * 32 - 1);
  bx[3] = (float)((c + rws[ri]) * 32 - 1);
}

__global__ __launch_bounds__(128) void appm_kernel(
    const float* __restrict__ att,   // [B][1][14][14]
    float* __restrict__ all_scores,  // [B][602]
    int*   __restrict__ indices,     // [B][7]
    float* __restrict__ win_scores,  // [B][7]
    int*   __restrict__ coords,      // [B][7][4]
    float* __restrict__ boxes_ws)    // [B][7][4]
{
  __shared__ float a[196];
  __shared__ float sc[602];
  __shared__ float orig[602];
  const int b = blockIdx.x;

  for (int i = threadIdx.x; i < 196; i += blockDim.x) a[i] = att[b * 196 + i];
  __syncthreads();

  const int rhs[6] = {4, 3, 5, 6, 5, 7};
  const int rws[6] = {4, 5, 3, 6, 7, 5};
  const int off[7] = {0, 121, 241, 361, 442, 522, 602};

  for (int w = threadIdx.x; w < 602; w += blockDim.x) {
    int ri = 0;
    while (ri < 5 && w >= off[ri + 1]) ri++;
    int loc = w - off[ri];
    int rh = rhs[ri], rw = rws[ri];
    int cols = 14 - rw + 1;
    int r = loc / cols, c = loc % cols;
    float s = 0.f;
    for (int dy = 0; dy < rh; ++dy)
      for (int dx = 0; dx < rw; ++dx) s += a[(r + dy) * 14 + (c + dx)];
    s /= (float)(rh * rw);
    sc[w] = s;
    orig[w] = s;
    all_scores[(long)b * 602 + w] = s;
  }
  __syncthreads();

  if (threadIdx.x == 0) {
    const int gs[2] = {0, 361}, ge[2] = {361, 602}, nl[2] = {4, 3};
    int oi = 0;
    for (int g = 0; g < 2; ++g) {
      for (int pick = 0; pick < nl[g]; ++pick) {
        int best = gs[g];
        float bv = sc[gs[g]];
        for (int w = gs[g] + 1; w < ge[g]; ++w)
          if (sc[w] > bv) { bv = sc[w]; best = w; }
        float bb[4];
        box_of(best, bb);
        float a1 = (bb[2] - bb[0] + 1.f) * (bb[3] - bb[1] + 1.f);
        for (int w = gs[g]; w < ge[g]; ++w) {
          float ob[4];
          box_of(w, ob);
          float x0 = fmaxf(bb[0], ob[0]), y0 = fmaxf(bb[1], ob[1]);
          float x1 = fminf(bb[2], ob[2]), y1 = fminf(bb[3], ob[3]);
          float inter = fmaxf(x1 - x0 + 1.f, 0.f) * fmaxf(y1 - y0 + 1.f, 0.f);
          float a2 = (ob[2] - ob[0] + 1.f) * (ob[3] - ob[1] + 1.f);
          float iou = inter / (a1 + a2 - inter);
          if (iou > 0.25f) sc[w] = -__builtin_inff();
        }
        sc[best] = -__builtin_inff();
        indices[b * 7 + oi] = best;
        win_scores[b * 7 + oi] = orig[best];
        for (int q = 0; q < 4; ++q) {
          coords[(b * 7 + oi) * 4 + q] = (int)bb[q];
          boxes_ws[(b * 7 + oi) * 4 + q] = bb[q];
        }
        oi++;
      }
    }
  }
}

// --------------------------- bilinear crop-resize --------------------------
// out: [112 proposals][112][112][3] f16 (NHWC, encoder input layout)
__global__ void crop_resize_kernel(const float* __restrict__ x,
                                   const float* __restrict__ boxes,
                                   _Float16* __restrict__ out) {
  const long total = (long)112 * 112 * 112 * 3;
  long idx = (long)blockIdx.x * blockDim.x + threadIdx.x;
  if (idx >= total) return;
  long t = idx;
  int ch = t % 3; t /= 3;
  int j = t % 112; t /= 112;
  int i = t % 112; t /= 112;
  int p = (int)t;            // proposal index 0..111
  int img = p / 7;
  const float* bx = boxes + p * 4;
  float ti = (float)i / 111.f;
  float tj = (float)j / 111.f;
  float rs = bx[0] + (bx[2] - bx[0]) * ti;
  float cs = bx[1] + (bx[3] - bx[1]) * tj;
  float r0f = floorf(rs), c0f = floorf(cs);
  float wr = rs - r0f, wc = cs - c0f;
  int r0 = (int)r0f, c0 = (int)c0f;
  int r1 = min(r0 + 1, 447), c1 = min(c0 + 1, 447);
  const float* im = x + ((long)img * 3 + ch) * 448 * 448;
  float g00 = im[(long)r0 * 448 + c0];
  float g01 = im[(long)r0 * 448 + c1];
  float g10 = im[(long)r1 * 448 + c0];
  float g11 = im[(long)r1 * 448 + c1];
  float top = (1.f - wc) * g00 + wc * g01;
  float bot = (1.f - wc) * g10 + wc * g11;
  out[idx] = (_Float16)((1.f - wr) * top + wr * bot);
}

// -------------------------------- launch -----------------------------------

extern "C" void kernel_launch(void* const* d_in, const int* in_sizes, int n_in,
                              void* d_out, int out_size, void* d_ws, size_t ws_size,
                              hipStream_t stream) {
  const float* x     = (const float*)d_in[0];
  const float* att   = (const float*)d_in[1];
  const float* cw[5] = {(const float*)d_in[2], (const float*)d_in[4],
                        (const float*)d_in[6], (const float*)d_in[8],
                        (const float*)d_in[10]};
  const float* cb[5] = {(const float*)d_in[3], (const float*)d_in[5],
                        (const float*)d_in[7], (const float*)d_in[9],
                        (const float*)d_in[11]};
  const float* head_w = (const float*)d_in[12];
  const float* head_b = (const float*)d_in[13];
  float* out = (float*)d_out;

  // output layout (floats/int32, concatenated in return order)
  const long OUT_LABELS = 0;       // 16*200
  const long OUT_WSC    = 3200;    // 16*7
  const long OUT_WLOG   = 3312;    // 112*200
  const long OUT_IDX    = 25712;   // 16*7 (int32)
  const long OUT_ALLSC  = 25824;   // 16*602
  const long OUT_COORD  = 35456;   // 16*7*4 (int32)

  // workspace layout (ping-pong f16 activation buffers + packed weights)
  char* ws = (char*)d_ws;
  _Float16* buf0  = (_Float16*)(ws);                        // <= 52 MB used
  _Float16* buf1  = (_Float16*)(ws + ((size_t)56 << 20));   // <= 103 MB used
  _Float16* wkbuf = (_Float16*)(ws + ((size_t)160 << 20));  // ~7.5 MB
  float*    boxes = (float*)   (ws + ((size_t)168 << 20));  // 448 floats

  const int CiL[5] = {3, 64, 128, 256, 512};
  const int CoL[5] = {64, 128, 256, 512, 512};
  const size_t wkOff[5] = {0, 1728, 75456, 370368, 1550016};

  // 1) pack weights f32 OIHW -> f16 [Co][K]
  for (int l = 0; l < 5; ++l) {
    long total = 9L * CiL[l] * CoL[l];
    pack_weights_f16<<<(unsigned)((total + 255) / 256), 256, 0, stream>>>(
        cw[l], wkbuf + wkOff[l], CoL[l], CiL[l]);
  }

  // 2) pack x f32 NCHW -> f16 NHWC into buf0
  {
    long total = 16L * 3 * 448 * 448;
    pack_nchw_to_nhwc_f16<<<(unsigned)((total + 255) / 256), 256, 0, stream>>>(
        x, buf0, 16, 3, 448, 448);
  }

#define LAUNCH_CONV(HI, CI, HO, CO, PAD, NB, PIN, WOFS, BI, POUT)              \
  do {                                                                         \
    dim3 g((((HO) * (HO)) + 63) / 64, (CO) / 64, (NB));                        \
    conv3x3_s2_wmma<HI, HI, CI, HO, HO, CO, PAD><<<g, 128, 0, stream>>>(       \
        (PIN), wkbuf + (WOFS), (BI), (POUT));                                  \
  } while (0)

  // 3) main encoder (B=16): 448->224->112->56->28->14
  LAUNCH_CONV(448,   3, 224,  64, 0, 16, buf0, wkOff[0], cb[0], buf1);
  LAUNCH_CONV(224,  64, 112, 128, 0, 16, buf1, wkOff[1], cb[1], buf0);
  LAUNCH_CONV(112, 128,  56, 256, 0, 16, buf0, wkOff[2], cb[2], buf1);
  LAUNCH_CONV( 56, 256,  28, 512, 0, 16, buf1, wkOff[3], cb[3], buf0);
  LAUNCH_CONV( 28, 512,  14, 512, 0, 16, buf0, wkOff[4], cb[4], buf1);
  // a5 (16x14x14x512) in buf1

  // 4) labels = head(pool(a5))
  pool_head<<<16, 256, 0, stream>>>(buf1, head_w, head_b, out + OUT_LABELS, 196);

  // 5) APPM: window scores + NMS -> indices, win_scores, coords, boxes
  appm_kernel<<<16, 128, 0, stream>>>(
      att, out + OUT_ALLSC, (int*)(out + OUT_IDX), out + OUT_WSC,
      (int*)(out + OUT_COORD), boxes);

  // 6) crop-resize proposals -> f16 NHWC encoder input (buf0)
  {
    long total = (long)112 * 112 * 112 * 3;
    crop_resize_kernel<<<(unsigned)((total + 255) / 256), 256, 0, stream>>>(
        x, boxes, buf0);
  }

  // 7) window encoder (B=112): 112->56->28->14->7->4
  LAUNCH_CONV(112,   3,  56,  64, 0, 112, buf0, wkOff[0], cb[0], buf1);
  LAUNCH_CONV( 56,  64,  28, 128, 0, 112, buf1, wkOff[1], cb[1], buf0);
  LAUNCH_CONV( 28, 128,  14, 256, 0, 112, buf0, wkOff[2], cb[2], buf1);
  LAUNCH_CONV( 14, 256,   7, 512, 0, 112, buf1, wkOff[3], cb[3], buf0);
  LAUNCH_CONV(  7, 512,   4, 512, 1, 112, buf0, wkOff[4], cb[4], buf1);
  // wa5 (112x4x4x512) in buf1

  // 8) win_logits = head(pool(wa5))
  pool_head<<<112, 256, 0, stream>>>(buf1, head_w, head_b, out + OUT_WLOG, 16);
}
